// Net_41807211660013
// MI455X (gfx1250) — matrix operand order repaired and checked by reading, not compiled, not run on hardware
//
#include <hip/hip_runtime.h>
#include <hip/hip_bf16.h>
#include <math.h>

#define NN   8192
#define EE   524288
#define FIN  39
#define HH   80
#define MSG  164
#define KPOST 1968          // 12 * MSG
#define LDK  2048           // KPOST padded so LDK % KC == 0
#define KC   128            // K-chunk staged in LDS per step
#define NCH  (LDK / KC)     // 16 chunks

typedef __attribute__((ext_vector_type(16))) __bf16 v16bf;
typedef __attribute__((ext_vector_type(8)))  __bf16 v8bf;
typedef __attribute__((ext_vector_type(8)))  float  v8f;
typedef int v4i __attribute__((vector_size(16)));

#ifndef __has_builtin
#define __has_builtin(x) 0
#endif
#if __has_builtin(__builtin_amdgcn_global_load_async_to_lds_b128)
#define HAVE_ASYNC_LDS 1
#else
#define HAVE_ASYNC_LDS 0
#endif

// ---------- helpers ----------
__device__ __forceinline__ unsigned f2ord(float f) {
    unsigned u = __float_as_uint(f);
    return (u & 0x80000000u) ? ~u : (u | 0x80000000u);
}
__device__ __forceinline__ float ord2f(unsigned o) {
    unsigned u = (o & 0x80000000u) ? (o ^ 0x80000000u) : ~o;
    return __uint_as_float(u);
}

__device__ __forceinline__ v8f wmma_bf16(v16bf a, v16bf b, v8f c) {
    return __builtin_amdgcn_wmma_f32_16x16x32_bf16(false, a, false, b, (short)0, c,
                                                   false, false);
}

// A fragment, 16x32 bf16: lane holds row; element i -> K = (i<8?0:16)+(half?8:0)+(i&7)
__device__ __forceinline__ v16bf load_frag_a(const __bf16* row, int k0, int half) {
    const __bf16* p = row + k0 + (half ? 8 : 0);
    v8bf lo = *(const v8bf*)(p);
    v8bf hi = *(const v8bf*)(p + 16);
    v16bf r;
#pragma unroll
    for (int i = 0; i < 8; ++i) { r[i] = lo[i]; r[i + 8] = hi[i]; }
    return r;
}
// B fragment, 32x16 bf16 (K-contiguous source row per output column): K = (i<8?0:16)+(half?8:0)+(i&7)
__device__ __forceinline__ v16bf load_frag_b(const __bf16* col, int k0, int half) {
    const __bf16* p = col + k0 + (half ? 8 : 0);
    v8bf lo = *(const v8bf*)(p);
    v8bf hi = *(const v8bf*)(p + 16);
    v16bf r;
#pragma unroll
    for (int i = 0; i < 8; ++i) { r[i] = lo[i]; r[i + 8] = hi[i]; }
    return r;
}

// 16B global -> LDS, async on CDNA5 (ASYNCcnt), sync fallback otherwise
__device__ __forceinline__ void stage16(const __bf16* g, __bf16* l) {
#if HAVE_ASYNC_LDS
    __builtin_amdgcn_global_load_async_to_lds_b128(
        (__attribute__((address_space(1))) v4i*)g,
        (__attribute__((address_space(3))) v4i*)l, 0, 0);
#else
    *(v8bf*)l = *(const v8bf*)g;
#endif
}

__device__ __forceinline__ void wait_stage() {
#if HAVE_ASYNC_LDS
#if __has_builtin(__builtin_amdgcn_s_wait_asynccnt)
    __builtin_amdgcn_s_wait_asynccnt(0);
#else
    asm volatile("s_wait_asynccnt 0x0" ::: "memory");
#endif
#endif
    __syncthreads();
}

// ---------- small kernels ----------
__global__ void k_zero_f(float* p, int n) {
    int g = blockIdx.x * 256 + threadIdx.x;
    if (g < n) p[g] = 0.0f;
}

__global__ void k_deg(const long long* __restrict__ ei, float* __restrict__ deg) {
    int e = blockIdx.x * 256 + threadIdx.x;
    if (e < EE) atomicAdd(&deg[(int)ei[EE + e]], 1.0f);
}

__global__ void k_avglog(const float* __restrict__ deg, float* __restrict__ avg) {
    __shared__ float sm[256];
    float a = 0.0f;
    for (int i = threadIdx.x; i < NN; i += 256) a += log1pf(deg[i]);
    sm[threadIdx.x] = a;
    __syncthreads();
    for (int s = 128; s > 0; s >>= 1) {
        if (threadIdx.x < s) sm[threadIdx.x] += sm[threadIdx.x + s];
        __syncthreads();
    }
    if (threadIdx.x == 0) *avg = sm[0] / (float)NN;
}

__global__ void k_scalers(const float* __restrict__ deg, const float* __restrict__ avg,
                          float* __restrict__ amp, float* __restrict__ att,
                          float* __restrict__ dsafe, float* __restrict__ hasf) {
    int n = blockIdx.x * 256 + threadIdx.x;
    if (n >= NN) return;
    float d = deg[n];
    float ld = log1pf(d);
    float av = *avg;
    amp[n] = ld / av;
    att[n] = (ld > 0.0f) ? (av / ld) : 1.0f;
    dsafe[n] = (d > 0.0f) ? d : 1.0f;
    hasf[n] = (d > 0.0f) ? 1.0f : 0.0f;
}

// generic dense: out[n,j] = act(sum_k in[n,k]*W[k,j] + b[j])
__global__ void k_dense(const float* __restrict__ in, const float* __restrict__ W,
                        const float* __restrict__ b, float* __restrict__ out,
                        int nrows, int kdim, int jdim, int dorelu) {
    int g = blockIdx.x * 256 + threadIdx.x;
    if (g >= nrows * jdim) return;
    int n = g / jdim, j = g - n * jdim;
    float a = b[j];
    for (int k = 0; k < kdim; ++k) a += in[n * kdim + k] * W[k * jdim + j];
    out[g] = dorelu ? fmaxf(a, 0.0f) : a;
}

// ---------- aggregation ----------
__global__ void k_agg_init(float* __restrict__ asum, float* __restrict__ assq,
                           unsigned* __restrict__ amn, unsigned* __restrict__ amx) {
    int g = blockIdx.x * 256 + threadIdx.x;   // exact grid: NN*MSG
    asum[g] = 0.0f;
    assq[g] = 0.0f;
    amn[g] = 0xFFFFFFFFu;
    amx[g] = 0x00000000u;
}

__global__ void k_agg(const float* __restrict__ h, const float* __restrict__ eattr,
                      const long long* __restrict__ ei,
                      float* __restrict__ asum, float* __restrict__ assq,
                      unsigned* __restrict__ amn, unsigned* __restrict__ amx) {
    int g = blockIdx.x * 256 + threadIdx.x;   // exact grid: EE*MSG
    int e = g / MSG;
    int c = g - e * MSG;
    int s = (int)ei[e];
    int d = (int)ei[EE + e];
    float v;
    if (c < HH)            v = h[d * HH + c];
    else if (c < 2 * HH)   v = h[s * HH + (c - HH)];
    else                   v = eattr[e * 4 + (c - 2 * HH)];
    int o = d * MSG + c;
    atomicAdd(&asum[o], v);
    atomicAdd(&assq[o], v * v);
    unsigned ov = f2ord(v);
    atomicMin(&amn[o], ov);
    atomicMax(&amx[o], ov);
}

// Build PNA feature matrix [NN, LDK] in bf16 (zero padded K)
__global__ void k_feat(const float* __restrict__ asum, const float* __restrict__ assq,
                       const unsigned* __restrict__ amn, const unsigned* __restrict__ amx,
                       const float* __restrict__ amp, const float* __restrict__ att,
                       const float* __restrict__ dsafe, const float* __restrict__ hasf,
                       __bf16* __restrict__ feat) {
    int g = blockIdx.x * 256 + threadIdx.x;   // exact grid: NN*LDK
    int n = g / LDK;
    int k = g - n * LDK;
    if (k >= KPOST) { feat[g] = (__bf16)0.0f; return; }
    int s = k / (4 * MSG);            // 0: identity, 1: amp, 2: att
    int r = k - s * (4 * MSG);
    int a = r / MSG;                  // 0 mean, 1 min, 2 max, 3 std
    int c = r - a * MSG;
    int o = n * MSG + c;
    float ds = dsafe[n];
    float mean = asum[o] / ds;
    float v;
    if (a == 0)      v = mean;
    else if (a == 1) v = (hasf[n] != 0.0f) ? ord2f(amn[o]) : 0.0f;
    else if (a == 2) v = (hasf[n] != 0.0f) ? ord2f(amx[o]) : 0.0f;
    else {
        float q = assq[o] / ds - mean * mean;
        v = sqrtf(fmaxf(q, 0.0f) + 1e-5f);
    }
    float sc = (s == 0) ? 1.0f : (s == 1 ? amp[n] : att[n]);
    feat[g] = (__bf16)(v * sc);
}

// transpose+convert W_post[l] (KPOST x HH) -> wbt (HH x LDK bf16, K-contiguous)
__global__ void k_wconv(const float* __restrict__ Wp, __bf16* __restrict__ wbt) {
    int g = blockIdx.x * 256 + threadIdx.x;   // exact grid: HH*LDK
    int j = g / LDK;
    int k = g - j * LDK;
    wbt[g] = (k < KPOST) ? (__bf16)Wp[k * HH + j] : (__bf16)0.0f;
}

// ---------- WMMA GEMM with async-staged B in LDS ----------
// Stage one K-chunk of B (all 80 columns x KC) into LDS: 80*128*2B = 20 KB.
__device__ __forceinline__ void stage_chunk(const __bf16* __restrict__ wbt,
                                            __bf16* dst, int k0, int tid) {
#pragma unroll
    for (int i = 0; i < (HH * KC / 8) / 256; ++i) {  // 1280 16B transfers / 256 thr = 5
        int t = i * 256 + tid;
        int col = t >> 4;
        int seg = t & 15;
        stage16(wbt + (size_t)col * LDK + k0 + seg * 8, dst + col * KC + seg * 8);
    }
}

__global__ void __launch_bounds__(256) k_gemm(const __bf16* __restrict__ feat,
                                              const __bf16* __restrict__ wbt,
                                              const float* __restrict__ bias,
                                              float* __restrict__ out) {
    __shared__ __bf16 lb[2][HH * KC];                // 2 x 20 KB of the 320 KB/WGP
    const int tid  = threadIdx.x;
    const int lane = tid & 31;
    const int wave = tid >> 5;
    const int half = lane >> 4;
    const int ll   = lane & 15;
    const int m0   = blockIdx.x * 128 + wave * 16;   // 64 blocks * 8 waves * 16 = 8192

    const __bf16* arow = feat + (size_t)(m0 + ll) * LDK;

    v8f acc[5];
#pragma unroll
    for (int t = 0; t < 5; ++t)
#pragma unroll
        for (int i = 0; i < 8; ++i) acc[t][i] = 0.0f;

    stage_chunk(wbt, &lb[0][0], 0, tid);             // prologue: chunk 0 -> buf 0

    for (int ch = 0; ch < NCH; ++ch) {
        wait_stage();                                // my ASYNCcnt==0, then block barrier
        if (ch + 1 < NCH)
            stage_chunk(wbt, &lb[(ch + 1) & 1][0], (ch + 1) * KC, tid);

        const __bf16* bbase = &lb[ch & 1][0];
        const int kg = ch * KC;
        __builtin_prefetch(arow + kg + KC, 0, 3);    // prefetch next A chunk
#pragma unroll
        for (int kk = 0; kk < KC; kk += 32) {
            v16bf A = load_frag_a(arow, kg + kk, half);
            v16bf B[5];
#pragma unroll
            for (int t = 0; t < 5; ++t)              // batch all ds_loads: one wait,
                B[t] = load_frag_b(bbase + (t * 16 + ll) * KC, kk, half);
#pragma unroll
            for (int t = 0; t < 5; ++t)              // then back-to-back WMMAs
                acc[t] = wmma_bf16(A, B[t], acc[t]);
        }
    }

#pragma unroll
    for (int t = 0; t < 5; ++t) {
        float bj = bias[t * 16 + ll];
#pragma unroll
        for (int v = 0; v < 8; ++v) {
            int row = m0 + v + half * 8;
            out[(size_t)row * HH + t * 16 + ll] = acc[t][v] + bj;
        }
    }
}

// ---------- batchnorm ----------
__global__ void k_bnstats(const float* __restrict__ out, float* __restrict__ mu,
                          float* __restrict__ var) {
    __shared__ float s1[256], s2[256];
    int j = blockIdx.x;                               // 80 blocks
    float a = 0.0f, b = 0.0f;
    for (int n = threadIdx.x; n < NN; n += 256) {
        float v = out[(size_t)n * HH + j];
        a += v; b += v * v;
    }
    s1[threadIdx.x] = a; s2[threadIdx.x] = b;
    __syncthreads();
    for (int s = 128; s > 0; s >>= 1) {
        if (threadIdx.x < s) { s1[threadIdx.x] += s1[threadIdx.x + s]; s2[threadIdx.x] += s2[threadIdx.x + s]; }
        __syncthreads();
    }
    if (threadIdx.x == 0) {
        float m = s1[0] / (float)NN;
        mu[j] = m;
        var[j] = s2[0] / (float)NN - m * m;
    }
}

__global__ void k_bnapply(const float* __restrict__ out, const float* __restrict__ mu,
                          const float* __restrict__ var, const float* __restrict__ gm,
                          const float* __restrict__ bt, float* __restrict__ h) {
    int g = blockIdx.x * 256 + threadIdx.x;           // exact grid: NN*HH
    int j = g % HH;
    float xh = (out[g] - mu[j]) * rsqrtf(var[j] + 1e-5f);
    h[g] += fmaxf(gm[j] * xh + bt[j], 0.0f);
}

// ---------- final projection prep ----------
__global__ void k_prep(const float* __restrict__ y, __bf16* __restrict__ yb,
                       float* __restrict__ sq) {
    int n = blockIdx.x * 256 + threadIdx.x;           // exact: 8192
    float a = 0.0f;
#pragma unroll
    for (int k = 0; k < 32; ++k) {
        float v = y[n * 32 + k];
        a += v * v;
        yb[n * 32 + k] = (__bf16)v;
    }
    sq[n] = a;
}

// ---------- pairwise UMAP similarity: one WMMA (K=32) per 16x16 tile ----------
__global__ void __launch_bounds__(256) k_pair(const __bf16* __restrict__ yb,
                                              const float* __restrict__ sq,
                                              float* __restrict__ out) {
    const int lane = threadIdx.x & 31;
    const int wave = threadIdx.x >> 5;
    const int half = lane >> 4;
    const int ll   = lane & 15;
    const int tile = blockIdx.x * 8 + wave;           // 32768 blocks * 8 = 512*512 tiles
    const int ti = tile >> 9;
    const int tj = tile & 511;
    const int i0 = ti * 16, j0 = tj * 16;

    v16bf A = load_frag_a(yb + (size_t)(i0 + ll) * 32, 0, half);
    v16bf B = load_frag_b(yb + (size_t)(j0 + ll) * 32, 0, half);
    v8f c;
#pragma unroll
    for (int i = 0; i < 8; ++i) c[i] = 0.0f;
    c = wmma_bf16(A, B, c);                           // c[v] = dot(y[i0+v+8h], y[j0+ll])

    const int j = j0 + ll;
    const float sqj = sq[j];
#pragma unroll
    for (int v = 0; v < 8; ++v) {
        int i = i0 + v + half * 8;
        float d2 = fmaxf(sq[i] + sqj - 2.0f * c[v], 0.0f);
        out[(size_t)i * NN + j] = 1.0f / (1.0f + 0.583f * __powf(d2, 1.334f));
    }
}

// ---------- launch ----------
extern "C" void kernel_launch(void* const* d_in, const int* in_sizes, int n_in,
                              void* d_out, int out_size, void* d_ws, size_t ws_size,
                              hipStream_t stream) {
    (void)in_sizes; (void)n_in; (void)out_size; (void)ws_size;
    const float* x      = (const float*)d_in[0];
    const float* eattr  = (const float*)d_in[1];
    const float* W_emb  = (const float*)d_in[2];
    const float* b_emb  = (const float*)d_in[3];
    const float* W_post = (const float*)d_in[4];
    const float* b_post = (const float*)d_in[5];
    const float* bn_g   = (const float*)d_in[6];
    const float* bn_b   = (const float*)d_in[7];
    const float* W1 = (const float*)d_in[8];
    const float* b1 = (const float*)d_in[9];
    const float* W2 = (const float*)d_in[10];
    const float* b2 = (const float*)d_in[11];
    const float* W3 = (const float*)d_in[12];
    const float* b3 = (const float*)d_in[13];
    const long long* ei = (const long long*)d_in[14];   // int64 per reference
    float* out = (float*)d_out;

    char* base = (char*)d_ws;
    size_t off = 0;
    auto alloc = [&](size_t bytes) -> void* {
        void* p = base + off;
        off = (off + bytes + 255) & ~(size_t)255;
        return p;
    };
    float*    deg   = (float*)alloc((size_t)NN * 4);
    float*    amp   = (float*)alloc((size_t)NN * 4);
    float*    att   = (float*)alloc((size_t)NN * 4);
    float*    dsafe = (float*)alloc((size_t)NN * 4);
    float*    hasf  = (float*)alloc((size_t)NN * 4);
    float*    avg   = (float*)alloc(256);
    float*    h     = (float*)alloc((size_t)NN * HH * 4);
    float*    asum  = (float*)alloc((size_t)NN * MSG * 4);
    float*    assq  = (float*)alloc((size_t)NN * MSG * 4);
    unsigned* amn   = (unsigned*)alloc((size_t)NN * MSG * 4);
    unsigned* amx   = (unsigned*)alloc((size_t)NN * MSG * 4);
    __bf16*   feat  = (__bf16*)alloc((size_t)NN * LDK * 2);
    __bf16*   wbt   = (__bf16*)alloc((size_t)HH * LDK * 2);
    float*    opost = (float*)alloc((size_t)NN * HH * 4);
    float*    mu    = (float*)alloc((size_t)HH * 4);
    float*    var   = (float*)alloc((size_t)HH * 4);
    float*    y1    = (float*)alloc((size_t)NN * 40 * 4);
    float*    y2    = (float*)alloc((size_t)NN * 20 * 4);
    float*    y3    = (float*)alloc((size_t)NN * 32 * 4);
    __bf16*   yb    = (__bf16*)alloc((size_t)NN * 32 * 2);
    float*    sq    = (float*)alloc((size_t)NN * 4);

    k_zero_f<<<NN / 256, 256, 0, stream>>>(deg, NN);
    k_deg<<<EE / 256, 256, 0, stream>>>(ei, deg);
    k_avglog<<<1, 256, 0, stream>>>(deg, avg);
    k_scalers<<<NN / 256, 256, 0, stream>>>(deg, avg, amp, att, dsafe, hasf);
    k_dense<<<(NN * HH) / 256, 256, 0, stream>>>(x, W_emb, b_emb, h, NN, FIN, HH, 0);

    for (int l = 0; l < 4; ++l) {
        k_agg_init<<<(NN * MSG) / 256, 256, 0, stream>>>(asum, assq, amn, amx);
        k_agg<<<(EE * MSG) / 256, 256, 0, stream>>>(h, eattr, ei, asum, assq, amn, amx);
        k_feat<<<(NN * LDK) / 256, 256, 0, stream>>>(asum, assq, amn, amx, amp, att,
                                                     dsafe, hasf, feat);
        k_wconv<<<(HH * LDK) / 256, 256, 0, stream>>>(W_post + (size_t)l * KPOST * HH, wbt);
        k_gemm<<<NN / 128, 256, 0, stream>>>(feat, wbt, b_post + l * HH, opost);
        k_bnstats<<<HH, 256, 0, stream>>>(opost, mu, var);
        k_bnapply<<<(NN * HH) / 256, 256, 0, stream>>>(opost, mu, var,
                                                       bn_g + l * HH, bn_b + l * HH, h);
    }

    k_dense<<<(NN * 40) / 256, 256, 0, stream>>>(h, W1, b1, y1, NN, HH, 40, 1);
    k_dense<<<(NN * 20) / 256, 256, 0, stream>>>(y1, W2, b2, y2, NN, 40, 20, 1);
    k_dense<<<(NN * 32) / 256, 256, 0, stream>>>(y2, W3, b3, y3, NN, 20, 32, 0);
    k_prep<<<NN / 256, 256, 0, stream>>>(y3, yb, sq);
    k_pair<<<(512 * 512) / 8, 256, 0, stream>>>(yb, sq, out);
}
